// L1RegressionActionHead_68513318306394
// MI455X (gfx1250) — compile-verified, matrix operand order checked by reading, not run on hardware
//
#include <hip/hip_runtime.h>
#include <cstdint>
#include <cstddef>

// ---------------------------------------------------------------------------
// Types for CDNA5 WMMA
// ---------------------------------------------------------------------------
typedef __bf16 bf16;
typedef __attribute__((ext_vector_type(16))) __bf16 v16bf;
typedef __attribute__((ext_vector_type(8)))  float  v8f;
typedef __attribute__((ext_vector_type(4)))  int    v4i;

// Address-space qualified int4-vector pointers for async global->LDS DMA
// (clang's builtin signature expects int __vector(4) pointers: AS1 src, AS3 dst)
typedef __attribute__((address_space(1))) v4i* as1_v4i_ptr;
typedef __attribute__((address_space(3))) v4i* as3_v4i_ptr;

#if defined(__has_builtin)
#if __has_builtin(__builtin_amdgcn_global_load_async_to_lds_b128) && \
    __has_builtin(__builtin_amdgcn_s_wait_asynccnt)
#define USE_ASYNC_LDS 1
#endif
#endif
#ifndef USE_ASYNC_LDS
#define USE_ASYNC_LDS 0
#endif

static __device__ __forceinline__ bf16 f2bf(float f) {
    union { float f; unsigned u; } a; a.f = f;
    unsigned r = a.u + 0x7FFFu + ((a.u >> 16) & 1u);   // round-to-nearest-even
    unsigned short h = (unsigned short)(r >> 16);
    return __builtin_bit_cast(bf16, h);
}

// ---------------------------------------------------------------------------
// Problem constants
// ---------------------------------------------------------------------------
#define DIMC 1024
#define HEADSC 8
#define HDC 128
#define BC 16
#define TC 1024
#define KTC 64
#define KAC 2
#define NKEYS 66

// ---------------------------------------------------------------------------
// Elementwise fp32 -> bf16 (4 elems / thread)
// ---------------------------------------------------------------------------
__global__ void cvt_f32_bf16(const float* __restrict__ in, bf16* __restrict__ out, int n) {
    int idx = (blockIdx.x * blockDim.x + threadIdx.x) * 4;
    if (idx + 3 < n) {
        float4 v = *reinterpret_cast<const float4*>(in + idx);
        out[idx + 0] = f2bf(v.x);
        out[idx + 1] = f2bf(v.y);
        out[idx + 2] = f2bf(v.z);
        out[idx + 3] = f2bf(v.w);
    } else {
        for (int j = idx; j < n; ++j) out[j] = f2bf(in[j]);
    }
}

// ---------------------------------------------------------------------------
// W (K x N, fp32, row-major) -> Wt (N x K, bf16, row-major)
// ---------------------------------------------------------------------------
__global__ void transpose_cvt(const float* __restrict__ W, bf16* __restrict__ Wt,
                              int K, int N) {
    __shared__ float tile[32][33];
    const int bx = blockIdx.x * 32;   // N base
    const int by = blockIdx.y * 32;   // K base
    const int tx = threadIdx.x;       // 0..31
    const int ty = threadIdx.y;       // 0..7
    for (int i = ty; i < 32; i += 8) {
        int k = by + i, n = bx + tx;
        tile[i][tx] = (k < K && n < N) ? W[(size_t)k * N + n] : 0.0f;
    }
    __syncthreads();
    for (int i = ty; i < 32; i += 8) {
        int n = bx + i, k = by + tx;
        if (n < N && k < K) Wt[(size_t)n * K + k] = f2bf(tile[tx][i]);
    }
}

// ---------------------------------------------------------------------------
// h_ad = concat(h_a, p) along seq axis -> bf16 (B, 2, DIM)
// ---------------------------------------------------------------------------
__global__ void build_h_ad(const float* __restrict__ h_a, const float* __restrict__ p,
                           bf16* __restrict__ out, int total) {
    int i = blockIdx.x * blockDim.x + threadIdx.x;
    if (i >= total) return;
    int d = i % DIMC;
    int r = (i / DIMC) & 1;
    int b = i / (2 * DIMC);
    float v = r ? p[(size_t)b * DIMC + d] : h_a[(size_t)b * DIMC + d];
    out[i] = f2bf(v);
}

// ---------------------------------------------------------------------------
// Tiled bf16 WMMA GEMM: C(MxN,f32) = A(MxK,bf16) * Bt(NxK,bf16)^T + bias [+relu]
// Block tile 128x128, 256 threads = 8 waves (wave32), wave tile 64x32
// (4x2 WMMA 16x16 subtiles), K step 32 == one v_wmma_f32_16x16x32_bf16.
// Double-buffered LDS pipeline: stage tile kt+1 while computing tile kt.
// OOB rows are CLAMPED (not zeroed): a garbage A-row m / B-row n only feeds
// C[m][*] / C[*][n], which the epilogue never stores -> branch-free staging.
// ---------------------------------------------------------------------------
__global__ __launch_bounds__(256)
void gemm_bf16_wmma(const bf16* __restrict__ A, const bf16* __restrict__ Bt,
                    const float* __restrict__ bias, float* __restrict__ C,
                    int M, int N, int K, int relu) {
    __shared__ alignas(16) bf16 As[2][128][40];   // 2 bufs x 128 rows x 32 K (pad 8)
    __shared__ alignas(16) bf16 Bs[2][128][40];

    const int tid  = threadIdx.x;
    const int lane = tid & 31;
    const int wid  = tid >> 5;
    const int wm   = wid >> 2;                 // 0..1 -> M offset wm*64
    const int wn   = wid & 3;                  // 0..3 -> N offset wn*32
    const int blockM = blockIdx.y * 128;
    const int blockN = blockIdx.x * 128;

    v8f acc[4][2];
    const v8f vz = {};
    #pragma unroll
    for (int i = 0; i < 4; ++i)
        #pragma unroll
        for (int j = 0; j < 2; ++j) acc[i][j] = vz;

    // LDS fill mapping: thread -> (row, 16-elem half); rows clamped in range.
    const int lrow = tid >> 1;                 // 0..127
    const int lcol = (tid & 1) * 16;           // 0 or 16
    int ar = blockM + lrow; if (ar >= M) ar = M - 1;
    int br = blockN + lrow; if (br >= N) br = N - 1;
    const bf16* aPtr = A  + (size_t)ar * K + lcol;
    const bf16* bPtr = Bt + (size_t)br * K + lcol;

    const int fr = lane & 15;                  // row/col within 16
    const int kh = (lane >> 4) * 8;            // K half select (ISA 16-bit layout)
    const int kTiles = K / 32;

#if USE_ASYNC_LDS
    // --- async DMA staging (GLOBAL_LOAD_ASYNC_TO_LDS_B128, ASYNCcnt) ---
    {
        const bf16* ga = aPtr;
        const bf16* gb = bPtr;
        __builtin_amdgcn_global_load_async_to_lds_b128(
            (as1_v4i_ptr)ga,       (as3_v4i_ptr)&As[0][lrow][lcol],     0, 0);
        __builtin_amdgcn_global_load_async_to_lds_b128(
            (as1_v4i_ptr)(ga + 8), (as3_v4i_ptr)&As[0][lrow][lcol + 8], 0, 0);
        __builtin_amdgcn_global_load_async_to_lds_b128(
            (as1_v4i_ptr)gb,       (as3_v4i_ptr)&Bs[0][lrow][lcol],     0, 0);
        __builtin_amdgcn_global_load_async_to_lds_b128(
            (as1_v4i_ptr)(gb + 8), (as3_v4i_ptr)&Bs[0][lrow][lcol + 8], 0, 0);
    }
    __builtin_amdgcn_s_wait_asynccnt(0);
    __syncthreads();
#else
    // --- register-staged prologue ---
    {
        const uint4* ga = reinterpret_cast<const uint4*>(aPtr);
        const uint4* gb = reinterpret_cast<const uint4*>(bPtr);
        uint4 a0 = ga[0], a1 = ga[1], b0 = gb[0], b1 = gb[1];
        *reinterpret_cast<uint4*>(&As[0][lrow][lcol + 0]) = a0;
        *reinterpret_cast<uint4*>(&As[0][lrow][lcol + 8]) = a1;
        *reinterpret_cast<uint4*>(&Bs[0][lrow][lcol + 0]) = b0;
        *reinterpret_cast<uint4*>(&Bs[0][lrow][lcol + 8]) = b1;
    }
    __syncthreads();
#endif

    for (int kt = 0; kt < kTiles; ++kt) {
        const int cur = kt & 1;
        const int nxt = cur ^ 1;
        const bool more = (kt + 1) < kTiles;

#if USE_ASYNC_LDS
        if (more) {   // stage tile kt+1 into other buffer while computing
            const bf16* ga = aPtr + (size_t)(kt + 1) * 32;
            const bf16* gb = bPtr + (size_t)(kt + 1) * 32;
            __builtin_amdgcn_global_load_async_to_lds_b128(
                (as1_v4i_ptr)ga,       (as3_v4i_ptr)&As[nxt][lrow][lcol],     0, 0);
            __builtin_amdgcn_global_load_async_to_lds_b128(
                (as1_v4i_ptr)(ga + 8), (as3_v4i_ptr)&As[nxt][lrow][lcol + 8], 0, 0);
            __builtin_amdgcn_global_load_async_to_lds_b128(
                (as1_v4i_ptr)gb,       (as3_v4i_ptr)&Bs[nxt][lrow][lcol],     0, 0);
            __builtin_amdgcn_global_load_async_to_lds_b128(
                (as1_v4i_ptr)(gb + 8), (as3_v4i_ptr)&Bs[nxt][lrow][lcol + 8], 0, 0);
        }
#else
        uint4 a0, a1, b0, b1;
        if (more) {   // issue global loads early; store to LDS after compute
            const uint4* ga = reinterpret_cast<const uint4*>(aPtr + (size_t)(kt + 1) * 32);
            const uint4* gb = reinterpret_cast<const uint4*>(bPtr + (size_t)(kt + 1) * 32);
            a0 = ga[0]; a1 = ga[1]; b0 = gb[0]; b1 = gb[1];
        }
        if (kt + 2 < kTiles) {   // near-scope prefetch two tiles ahead
            __builtin_prefetch(aPtr + (size_t)(kt + 2) * 32, 0, 3);
            __builtin_prefetch(bPtr + (size_t)(kt + 2) * 32, 0, 3);
        }
#endif

        // ---- fragments (per-lane K in {kh..kh+7} U {16+kh..16+kh+7}) ----
        v16bf afrag[4], bfrag[2];
        #pragma unroll
        for (int ms = 0; ms < 4; ++ms) {
            const int r = wm * 64 + ms * 16 + fr;
            union { v16bf v; uint4 u[2]; } f;
            f.u[0] = *reinterpret_cast<const uint4*>(&As[cur][r][kh]);
            f.u[1] = *reinterpret_cast<const uint4*>(&As[cur][r][kh + 16]);
            afrag[ms] = f.v;
        }
        #pragma unroll
        for (int ns = 0; ns < 2; ++ns) {
            const int r = wn * 32 + ns * 16 + fr;
            union { v16bf v; uint4 u[2]; } f;
            f.u[0] = *reinterpret_cast<const uint4*>(&Bs[cur][r][kh]);
            f.u[1] = *reinterpret_cast<const uint4*>(&Bs[cur][r][kh + 16]);
            bfrag[ns] = f.v;
        }

        #pragma unroll
        for (int ms = 0; ms < 4; ++ms)
            #pragma unroll
            for (int ns = 0; ns < 2; ++ns)
                acc[ms][ns] = __builtin_amdgcn_wmma_f32_16x16x32_bf16(
                    false, afrag[ms], false, bfrag[ns],
                    (short)0, acc[ms][ns], false, false);

#if USE_ASYNC_LDS
        __builtin_amdgcn_s_wait_asynccnt(0);
#else
        if (more) {
            *reinterpret_cast<uint4*>(&As[nxt][lrow][lcol + 0]) = a0;
            *reinterpret_cast<uint4*>(&As[nxt][lrow][lcol + 8]) = a1;
            *reinterpret_cast<uint4*>(&Bs[nxt][lrow][lcol + 0]) = b0;
            *reinterpret_cast<uint4*>(&Bs[nxt][lrow][lcol + 8]) = b1;
        }
#endif
        __syncthreads();
    }

    // ---- epilogue: C fragment row = r + (lane>=16)*8, col = lane&15 ----
    #pragma unroll
    for (int ms = 0; ms < 4; ++ms) {
        #pragma unroll
        for (int ns = 0; ns < 2; ++ns) {
            const int col  = blockN + wn * 32 + ns * 16 + (lane & 15);
            const int row0 = blockM + wm * 64 + ms * 16 + ((lane >> 4) * 8);
            if (col < N) {
                const float bv = bias ? bias[col] : 0.0f;
                #pragma unroll
                for (int r = 0; r < 8; ++r) {
                    const int row = row0 + r;
                    if (row < M) {
                        float v = acc[ms][ns][r] + bv;
                        if (relu) v = fmaxf(v, 0.0f);
                        C[(size_t)row * N + col] = v;
                    }
                }
            }
        }
    }
}

// ---------------------------------------------------------------------------
// RoPE in-place on (rows, DIM) fp32; pos = row % L; per-element angle
// t * 10000^{-(j mod 64)/64} with interleaved rotate-half (matches reference).
// One thread per pair -> no write races.
// ---------------------------------------------------------------------------
__global__ void rope_inplace(float* __restrict__ buf, int rows, int L) {
    const int pairsPerRow = DIMC / 2;
    int pi = blockIdx.x * blockDim.x + threadIdx.x;
    if (pi >= rows * pairsPerRow) return;
    const int row = pi / pairsPerRow;
    const int d0  = (pi % pairsPerRow) * 2;
    const int t   = row % L;
    const int dd0 = d0 & (HDC - 1);
    const int f0  = dd0 & 63;
    const int f1  = (dd0 + 1) & 63;
    const float inv0 = __powf(10000.0f, -(float)f0 / 64.0f);
    const float inv1 = __powf(10000.0f, -(float)f1 / 64.0f);
    float s0, c0, s1, c1;
    __sincosf((float)t * inv0, &s0, &c0);
    __sincosf((float)t * inv1, &s1, &c1);
    const size_t base = (size_t)row * DIMC + d0;
    const float x0 = buf[base], x1 = buf[base + 1];
    buf[base]     = x0 * c0 - x1 * s0;   // even: x*cos - x_odd*sin
    buf[base + 1] = x1 * c1 + x0 * s1;   // odd : x*cos + x_even*sin
}

// ---------------------------------------------------------------------------
// Attention: one wave32 per query (b,h,t). 66 keys = 2 adapter + 64 task.
// Lane l owns dims {l, l+32, l+64, l+96} of HD=128.
// ---------------------------------------------------------------------------
__global__ __launch_bounds__(256)
void attention_kernel(const float* __restrict__ q_a, const float* __restrict__ q_t,
                      const float* __restrict__ k_a, const float* __restrict__ k_t,
                      const float* __restrict__ v_a, const float* __restrict__ v_t,
                      const float* __restrict__ g, float* __restrict__ out) {
    __shared__ float sc[8][72];
    const int lane = threadIdx.x & 31;
    const int wid  = threadIdx.x >> 5;
    const int qid  = blockIdx.x * 8 + wid;
    const int b    = qid / (HEADSC * TC);
    const int rem  = qid % (HEADSC * TC);
    const int h    = rem / TC;
    const int t    = rem % TC;
    const float ratio  = 1.0f / (1.0f + __expf(-g[0]));
    const float rscale = 0.08838834764831845f;           // 1/sqrt(128)

    const size_t qoff = ((size_t)b * TC + t) * DIMC + h * HDC;
    float qa[4], qt[4];
    #pragma unroll
    for (int i = 0; i < 4; ++i) {
        qa[i] = q_a[qoff + lane + 32 * i];
        qt[i] = q_t[qoff + lane + 32 * i];
    }

    // scores
    for (int j = 0; j < NKEYS; ++j) {
        const float* kp;
        const float* qf;
        float mult;
        if (j < KAC) { kp = k_a + ((size_t)b * KAC + j)        * DIMC + h * HDC; qf = qa; mult = rscale; }
        else         { kp = k_t + ((size_t)b * KTC + (j - KAC)) * DIMC + h * HDC; qf = qt; mult = rscale * ratio; }
        float pd = 0.0f;
        #pragma unroll
        for (int i = 0; i < 4; ++i) pd += qf[i] * kp[lane + 32 * i];
        for (int off = 16; off > 0; off >>= 1) pd += __shfl_xor(pd, off, 32);
        if (lane == 0) sc[wid][j] = pd * mult;
    }
    __syncthreads();

    // softmax over 66 (3 values per lane, 3rd only lanes 0..1)
    float s0 = sc[wid][lane];
    float s1 = sc[wid][lane + 32];
    float s2 = (lane < 2) ? sc[wid][lane + 64] : -3.0e38f;
    float m = fmaxf(fmaxf(s0, s1), s2);
    for (int off = 16; off > 0; off >>= 1) m = fmaxf(m, __shfl_xor(m, off, 32));
    float e0 = __expf(s0 - m), e1 = __expf(s1 - m);
    float e2 = (lane < 2) ? __expf(s2 - m) : 0.0f;
    float sum = e0 + e1 + e2;
    for (int off = 16; off > 0; off >>= 1) sum += __shfl_xor(sum, off, 32);
    const float rs = 1.0f / sum;
    sc[wid][lane]      = e0 * rs;
    sc[wid][lane + 32] = e1 * rs;
    if (lane < 2) sc[wid][lane + 64] = e2 * rs;
    __syncthreads();

    // attn @ V
    float acc[4] = {0.f, 0.f, 0.f, 0.f};
    for (int j = 0; j < NKEYS; ++j) {
        const float* vp = (j < KAC)
            ? (v_a + ((size_t)b * KAC + j)         * DIMC + h * HDC)
            : (v_t + ((size_t)b * KTC + (j - KAC)) * DIMC + h * HDC);
        const float w = sc[wid][j];
        #pragma unroll
        for (int i = 0; i < 4; ++i) acc[i] += w * vp[lane + 32 * i];
    }
    #pragma unroll
    for (int i = 0; i < 4; ++i) out[qoff + lane + 32 * i] = acc[i];
}

// ---------------------------------------------------------------------------
// y = o + x; LayerNorm(y)*ln_g + ln_b -> bf16   (one block per row of 1024)
// ---------------------------------------------------------------------------
__global__ __launch_bounds__(256)
void residual_ln(const float* __restrict__ o, const float* __restrict__ x,
                 const float* __restrict__ gamma, const float* __restrict__ beta,
                 bf16* __restrict__ yn) {
    __shared__ float red[16];
    const int row = blockIdx.x;
    const int tid = threadIdx.x;
    const size_t base = (size_t)row * DIMC;
    float y[4];
    float s = 0.f, s2 = 0.f;
    #pragma unroll
    for (int i = 0; i < 4; ++i) {
        const int d = tid + 256 * i;
        const float v = o[base + d] + x[base + d];
        y[i] = v; s += v; s2 += v * v;
    }
    for (int off = 16; off > 0; off >>= 1) {
        s  += __shfl_xor(s,  off, 32);
        s2 += __shfl_xor(s2, off, 32);
    }
    const int lane = tid & 31, wid = tid >> 5;
    if (lane == 0) { red[wid] = s; red[8 + wid] = s2; }
    __syncthreads();
    float ts = 0.f, ts2 = 0.f;
    #pragma unroll
    for (int w = 0; w < 8; ++w) { ts += red[w]; ts2 += red[8 + w]; }
    const float mu   = ts * (1.0f / 1024.0f);
    const float var  = ts2 * (1.0f / 1024.0f) - mu * mu;
    const float rstd = rsqrtf(var + 1e-5f);
    #pragma unroll
    for (int i = 0; i < 4; ++i) {
        const int d = tid + 256 * i;
        yn[base + d] = f2bf((y[i] - mu) * rstd * gamma[d] + beta[d]);
    }
}

// ---------------------------------------------------------------------------
// Host orchestration
// ---------------------------------------------------------------------------
extern "C" void kernel_launch(void* const* d_in, const int* in_sizes, int n_in,
                              void* d_out, int out_size, void* d_ws, size_t ws_size,
                              hipStream_t stream) {
    (void)in_sizes; (void)n_in; (void)out_size; (void)ws_size;

    const float* x    = (const float*)d_in[0];
    const float* h_a  = (const float*)d_in[1];
    const float* h_t  = (const float*)d_in[2];
    const float* p    = (const float*)d_in[3];
    const float* w_qa = (const float*)d_in[4];  const float* b_qa = (const float*)d_in[5];
    const float* w_ka = (const float*)d_in[6];  const float* b_ka = (const float*)d_in[7];
    const float* w_va = (const float*)d_in[8];  const float* b_va = (const float*)d_in[9];
    const float* w_qt = (const float*)d_in[10]; const float* b_qt = (const float*)d_in[11];
    const float* w_kt = (const float*)d_in[12]; const float* b_kt = (const float*)d_in[13];
    const float* w_vt = (const float*)d_in[14]; const float* b_vt = (const float*)d_in[15];
    const float* w_o  = (const float*)d_in[16]; const float* b_o  = (const float*)d_in[17];
    const float* g    = (const float*)d_in[18];
    const float* ln_g = (const float*)d_in[19]; const float* ln_b = (const float*)d_in[20];
    const float* w_ffn= (const float*)d_in[21]; const float* b_ffn= (const float*)d_in[22];

    const int Mrows = BC * TC;                 // 16384
    const size_t nX   = (size_t)Mrows * DIMC;  // 16,777,216
    const size_t nHT  = (size_t)BC * KTC * DIMC;
    const size_t nHAD = (size_t)BC * KAC * DIMC;
    const size_t nW   = (size_t)DIMC * DIMC;

    // ---- workspace carve (aligned to 256B) ----
    char* ws = (char*)d_ws;
    size_t off = 0;
    auto carve = [&](size_t bytes) -> char* {
        char* ptr = ws + off;
        off += (bytes + 255) & ~(size_t)255;
        return ptr;
    };
    bf16*  xb    = (bf16*)carve(nX * sizeof(bf16));         // also reused as attn_out bf16
    bf16*  wt[8];
    for (int i = 0; i < 8; ++i) wt[i] = (bf16*)carve(nW * sizeof(bf16));
    bf16*  hadb  = (bf16*)carve(nHAD * sizeof(bf16));
    bf16*  htb   = (bf16*)carve(nHT * sizeof(bf16));
    float* q_a   = (float*)carve(nX * sizeof(float));       // reused as o_proj
    float* q_t   = (float*)carve(nX * sizeof(float));       // reused as yn bf16
    float* k_a   = (float*)carve(nHAD * sizeof(float));
    float* v_a   = (float*)carve(nHAD * sizeof(float));
    float* k_t   = (float*)carve(nHT * sizeof(float));
    float* v_t   = (float*)carve(nHT * sizeof(float));
    float* att   = (float*)carve(nX * sizeof(float));

    auto gemm = [&](const bf16* A, const bf16* Bt, const float* bias, float* C,
                    int M, int N, int K, int relu) {
        dim3 grid(N / 128, (M + 127) / 128);
        gemm_bf16_wmma<<<grid, 256, 0, stream>>>(A, Bt, bias, C, M, N, K, relu);
    };

    // 1) precision conversion / layout prep
    cvt_f32_bf16<<<(int)(nX / 1024), 256, 0, stream>>>(x, xb, (int)nX);
    cvt_f32_bf16<<<(int)(nHT / 1024), 256, 0, stream>>>(h_t, htb, (int)nHT);
    build_h_ad<<<(int)((nHAD + 255) / 256), 256, 0, stream>>>(h_a, p, hadb, (int)nHAD);
    const float* Wsrc[8] = {w_qa, w_qt, w_ka, w_va, w_kt, w_vt, w_o, w_ffn};
    for (int i = 0; i < 8; ++i)
        transpose_cvt<<<dim3(32, 32), dim3(32, 8), 0, stream>>>(Wsrc[i], wt[i], DIMC, DIMC);

    // 2) projections (WMMA bf16)
    gemm(xb,   wt[0], b_qa, q_a, Mrows,      DIMC, DIMC, 0);
    gemm(xb,   wt[1], b_qt, q_t, Mrows,      DIMC, DIMC, 0);
    gemm(hadb, wt[2], b_ka, k_a, BC * KAC,   DIMC, DIMC, 0);
    gemm(hadb, wt[3], b_va, v_a, BC * KAC,   DIMC, DIMC, 0);
    gemm(htb,  wt[4], b_kt, k_t, BC * KTC,   DIMC, DIMC, 0);
    gemm(htb,  wt[5], b_vt, v_t, BC * KTC,   DIMC, DIMC, 0);

    // 3) RoPE (in place; one thread per pair)
    rope_inplace<<<(Mrows * 512) / 256, 256, 0, stream>>>(q_a, Mrows, TC);
    rope_inplace<<<(Mrows * 512) / 256, 256, 0, stream>>>(q_t, Mrows, TC);
    rope_inplace<<<(BC * KAC * 512) / 256, 256, 0, stream>>>(k_a, BC * KAC, KAC);
    rope_inplace<<<(BC * KTC * 512) / 256, 256, 0, stream>>>(k_t, BC * KTC, KTC);

    // 4) attention (wave per query; 8 queries per block)
    attention_kernel<<<(BC * HEADSC * TC) / 8, 256, 0, stream>>>(
        q_a, q_t, k_a, k_t, v_a, v_t, g, att);

    // 5) output projection (reuse xb for bf16 attn_out, q_a for o_proj)
    bf16*  attb  = xb;
    float* oproj = q_a;
    cvt_f32_bf16<<<(int)(nX / 1024), 256, 0, stream>>>(att, attb, (int)nX);
    gemm(attb, wt[6], b_o, oproj, Mrows, DIMC, DIMC, 0);

    // 6) residual + LayerNorm -> bf16 (reuse q_t storage)
    bf16* ynb = (bf16*)q_t;
    residual_ln<<<Mrows, 256, 0, stream>>>(oproj, x, ln_g, ln_b, ynb);

    // 7) FFN GEMM with bias + ReLU -> d_out (fp32)
    gemm(ynb, wt[7], b_ffn, (float*)d_out, Mrows, DIMC, DIMC, 1);
}
